// EMAQuantize_55490977465091
// MI455X (gfx1250) — matrix operand order repaired and checked by reading, not compiled, not run on hardware
//
#include <hip/hip_runtime.h>
#include <hip/hip_bf16.h>
#include <math.h>

// ---------------------------------------------------------------------------
// VQ-VAE EMA quantizer for MI455X (gfx1250, wave32, WMMA).
// Core: fused [N,C]x[C,K] GEMM + argmin using v_wmma_f32_16x16x32_bf16 with
// double-buffered async global->LDS staging of the codebook tiles.
// N = 16*32*32 = 16384, C = 256, K = 8192.
// ---------------------------------------------------------------------------

#define NB   16
#define CH   256
#define HW   1024            // 32*32
#define NTOK 16384           // NB*HW
#define KCODE 8192
#define KT   64              // codes per LDS tile (64*256*2B = 32 KB)
#define DECAY 0.99f
#define ONE_M_DECAY 0.01f
#define COMMIT 0.25f
#define EPSV 1e-5f

typedef __bf16 bf16;
typedef __attribute__((ext_vector_type(16))) __bf16 v16bf;
typedef __attribute__((ext_vector_type(8)))  float  v8f;
typedef __attribute__((ext_vector_type(4)))  int    v4i;

typedef __attribute__((address_space(1))) v4i* gv4i_p;   // global int4*
typedef __attribute__((address_space(3))) v4i* lv4i_p;   // LDS int4*

union FragBF {
    v16bf v;
    uint4 u[2];
};

// ---- output layout (floats, concatenated in reference return order) -------
#define OFF_ZQ   0                         // 4194304
#define OFF_LOSS 4194304                   // 1
#define OFF_IDX  4194305                   // 16384
#define OFF_PERP 4210689                   // 1
#define OFF_NW   4210690                   // 2097152
#define OFF_NCS  6307842                   // 8192
#define OFF_NEA  6316034                   // 2097152

// ---------------------------------------------------------------------------
__global__ void vq_zero_ws(float* wsf, int nzero) {
    int i = blockIdx.x * blockDim.x + threadIdx.x;
    if (i < nzero) wsf[i] = 0.0f;
}

// one block (256 threads) per code row: bf16 convert + |w|^2
__global__ void vq_prep_w(const float* __restrict__ w, bf16* __restrict__ wb,
                          float* __restrict__ wnorm) {
    __shared__ float s[256];
    int k = blockIdx.x;
    int c = threadIdx.x;
    float v = w[(size_t)k * CH + c];
    wb[(size_t)k * CH + c] = (bf16)v;
    s[c] = v * v;
    __syncthreads();
    for (int off = 128; off > 0; off >>= 1) {
        if (c < off) s[c] += s[c + off];
        __syncthreads();
    }
    if (c == 0) wnorm[k] = s[0];
}

// z [B,C,H,W] f32 -> zb [N,C] bf16  (n = b*HW + hw)
__global__ void vq_prep_z(const float* __restrict__ z, bf16* __restrict__ zb) {
    int i = blockIdx.x * blockDim.x + threadIdx.x;     // linear over z
    if (i >= NB * CH * HW) return;
    int b  = i / (CH * HW);
    int r  = i - b * (CH * HW);
    int c  = r / HW;
    int hw = r - c * HW;
    int n  = b * HW + hw;
    zb[(size_t)n * CH + c] = (bf16)z[i];
}

// ---------------------------------------------------------------------------
// Async global->LDS prefetch of one KT-code tile (32 KB), 16 B per lane,
// 8 issues per thread. Tracked by ASYNCcnt (gfx1250 path).
// ---------------------------------------------------------------------------
__device__ __forceinline__ void tile_prefetch(const bf16* __restrict__ wb,
                                              int kbase, bf16* lds, int tid) {
    const uint4* g = reinterpret_cast<const uint4*>(wb + (size_t)kbase * CH);
    uint4* l = reinterpret_cast<uint4*>(lds);
#if __has_builtin(__builtin_amdgcn_global_load_async_to_lds_b128)
    #pragma unroll
    for (int r = 0; r < 8; ++r) {
        int chunk = tid + r * 256;
        __builtin_amdgcn_global_load_async_to_lds_b128(
            (gv4i_p)(g + chunk), (lv4i_p)(l + chunk), 0, 0);
    }
#else
    #pragma unroll
    for (int r = 0; r < 8; ++r) {
        int chunk = tid + r * 256;
        l[chunk] = g[chunk];
    }
#endif
}

__device__ __forceinline__ void wait_async_done() {
#if __has_builtin(__builtin_amdgcn_s_wait_asynccnt)
    __builtin_amdgcn_s_wait_asynccnt(0);
#else
    asm volatile("s_wait_asynccnt 0x0" ::: "memory");
#endif
}

// ---------------------------------------------------------------------------
// Fused GEMM + argmin. Each wave owns 16 z-rows (A pinned in VGPRs); the
// workgroup cooperatively stages KT codes of the codebook per tile into LDS
// (double-buffered, async), then each wave runs 8 WMMAs per 16-code chunk.
// ---------------------------------------------------------------------------
__global__ void __launch_bounds__(256)
vq_argmin_wmma(const bf16* __restrict__ zb, const bf16* __restrict__ wb,
               const float* __restrict__ wnorm, int* __restrict__ idx) {
    __shared__ __align__(16) bf16 smem[2][KT * CH];   // 2 x 32 KB

    const int tid  = threadIdx.x;
    const int lane = tid & 31;
    const int wave = tid >> 5;
    const int n0   = (blockIdx.x * (blockDim.x >> 5) + wave) * 16;
    const int col  = lane & 15;
    const bool hi  = lane >= 16;

    // Preload A fragments: row M = n0+col, 8 chunks of K=32 across C=256.
    // 16-bit A layout: lanes 0-15 K={c0..c0+7, c0+16..c0+23},
    //                  lanes 16-31 K={c0+8..c0+15, c0+24..c0+31}.
    FragBF a[8];
    const bf16* zrow = zb + (size_t)(n0 + col) * CH;
    #pragma unroll
    for (int cc = 0; cc < 8; ++cc) {
        int c0 = cc * 32 + (hi ? 8 : 0);
        a[cc].u[0] = *reinterpret_cast<const uint4*>(zrow + c0);
        a[cc].u[1] = *reinterpret_cast<const uint4*>(zrow + c0 + 16);
    }

    float minv[8];
    int   mini[8];
    #pragma unroll
    for (int j = 0; j < 8; ++j) { minv[j] = 3.4e38f; mini[j] = 0; }

    const int NTILES = KCODE / KT;        // 128
    tile_prefetch(wb, 0, smem[0], tid);

    for (int t = 0; t < NTILES; ++t) {
        wait_async_done();                // tile t resident (this wave's slice)
        __syncthreads();                  // all slices in + prev compute done
        if (t + 1 < NTILES)
            tile_prefetch(wb, (t + 1) * KT, smem[(t + 1) & 1], tid);

        const bf16* bufw = smem[t & 1];
        const int kbase = t * KT;
        #pragma unroll
        for (int q = 0; q < KT / 16; ++q) {
            v8f acc = {0.f, 0.f, 0.f, 0.f, 0.f, 0.f, 0.f, 0.f};
            // B column N = local code q*16+col; lanes 0-15 hold K={c0..c0+7, c0+16..c0+23},
            // lanes 16-31 hold K={c0+8..c0+15, c0+24..c0+31} (matches A layout).
            const bf16* wrow = bufw + (q * 16 + col) * CH;
            #pragma unroll
            for (int cc = 0; cc < 8; ++cc) {
                FragBF bfr;
                int c0 = cc * 32 + (hi ? 8 : 0);
                bfr.u[0] = *reinterpret_cast<const uint4*>(wrow + c0);
                bfr.u[1] = *reinterpret_cast<const uint4*>(wrow + c0 + 16);
                acc = __builtin_amdgcn_wmma_f32_16x16x32_bf16(
                    false, a[cc].v, false, bfr.v, (short)0, acc, false, false);
            }
            // dist' = |w_k|^2 - 2 * (z . w_k); |z|^2 constant per row.
            int   code = kbase + q * 16 + col;
            float wn   = wnorm[code];
            #pragma unroll
            for (int j = 0; j < 8; ++j) {
                float d = __builtin_fmaf(-2.0f, acc[j], wn);
                if (d < minv[j]) { minv[j] = d; mini[j] = code; }  // low k wins ties
            }
        }
    }

    // Reduce across the 16 columns held by each 16-lane half (xor masks 1..8
    // never cross bit 4, so halves stay independent under wave32 shuffles).
    #pragma unroll
    for (int j = 0; j < 8; ++j) {
        float v = minv[j];
        int   i = mini[j];
        #pragma unroll
        for (int m = 1; m < 16; m <<= 1) {
            float ov = __shfl_xor(v, m, 32);
            int   oi = __shfl_xor(i, m, 32);
            if (ov < v || (ov == v && oi < i)) { v = ov; i = oi; }
        }
        minv[j] = v; mini[j] = i;
    }
    if (col == 0) {
        int base = n0 + (hi ? 8 : 0);   // C/D layout: hi lanes hold M=8..15
        #pragma unroll
        for (int j = 0; j < 8; ++j) idx[base + j] = mini[j];
    }
}

// ---------------------------------------------------------------------------
// gather z_q (old weights), write [B,C,H,W], accumulate commitment loss.
__global__ void vq_gather_loss(const float* __restrict__ z,
                               const float* __restrict__ weight,
                               const int* __restrict__ idx,
                               float* __restrict__ zq_out,
                               float* __restrict__ loss_sum) {
    __shared__ float s[256];
    int i = blockIdx.x * blockDim.x + threadIdx.x;     // linear over z layout
    float d2 = 0.0f;
    if (i < NB * CH * HW) {
        int b  = i / (CH * HW);
        int r  = i - b * (CH * HW);
        int c  = r / HW;
        int hw = r - c * HW;
        int n  = b * HW + hw;
        int k  = idx[n];
        float q  = weight[(size_t)k * CH + c];
        float zz = z[i];
        zq_out[i] = q;                // STE forward value == z_q
        float dd = q - zz;
        d2 = dd * dd;
    }
    s[threadIdx.x] = d2;
    __syncthreads();
    for (int off = 128; off > 0; off >>= 1) {
        if (threadIdx.x < off) s[threadIdx.x] += s[threadIdx.x + off];
        __syncthreads();
    }
    if (threadIdx.x == 0) atomicAdd(loss_sum, s[0]);
}

__global__ void vq_idx_counts(const int* __restrict__ idx,
                              float* __restrict__ idx_out,
                              float* __restrict__ counts) {
    int n = blockIdx.x * blockDim.x + threadIdx.x;
    if (n >= NTOK) return;
    int k = idx[n];
    idx_out[n] = (float)k;
    atomicAdd(&counts[k], 1.0f);
}

// embed_sum[k, c] += z_flat[n, c]
__global__ void vq_embed_scatter(const float* __restrict__ z,
                                 const int* __restrict__ idx,
                                 float* __restrict__ embed_sum) {
    int i = blockIdx.x * blockDim.x + threadIdx.x;     // over N*C
    if (i >= NTOK * CH) return;
    int n = i / CH;
    int c = i - n * CH;
    int b  = n / HW;
    int hw = n - b * HW;
    int k = idx[n];
    float v = z[((size_t)b * CH + c) * HW + hw];
    atomicAdd(&embed_sum[(size_t)k * CH + c], v);
}

// per-code EMA of cluster size + entropy / n_total accumulation
__global__ void vq_ema_k(const float* __restrict__ cluster_size,
                         const float* __restrict__ counts,
                         float* __restrict__ ncs_out,
                         float* __restrict__ scal) {
    int k = blockIdx.x * blockDim.x + threadIdx.x;
    if (k >= KCODE) return;
    float cnt = counts[k];
    float ncs = DECAY * cluster_size[k] + ONE_M_DECAY * cnt;
    ncs_out[k] = ncs;
    atomicAdd(&scal[2], ncs);                                   // n_total
    float au = cnt / (float)NTOK;
    atomicAdd(&scal[1], au * logf(au + 1e-10f));                // entropy term
}

// per-(k,c) EMA of embed_avg
__global__ void vq_ema_kc(const float* __restrict__ embed_avg,
                          const float* __restrict__ embed_sum,
                          float* __restrict__ nea_out) {
    int i = blockIdx.x * blockDim.x + threadIdx.x;
    if (i >= KCODE * CH) return;
    nea_out[i] = DECAY * embed_avg[i] + ONE_M_DECAY * embed_sum[i];
}

__global__ void vq_finalize(const float* __restrict__ scal,
                            float* __restrict__ loss_out,
                            float* __restrict__ perp_out) {
    loss_out[0] = COMMIT * scal[0] / (float)(NTOK * CH);
    perp_out[0] = expf(-scal[1]);
}

// new_weight = new_embed_avg / smoothed(new_cluster_size)
__global__ void vq_new_weight(const float* __restrict__ nea,
                              const float* __restrict__ ncs,
                              const float* __restrict__ scal,
                              float* __restrict__ nw_out) {
    int i = blockIdx.x * blockDim.x + threadIdx.x;
    if (i >= KCODE * CH) return;
    int k = i / CH;
    float ntot = scal[2];
    float sm = (ncs[k] + EPSV) / (ntot + (float)KCODE * EPSV) * ntot;
    nw_out[i] = nea[i] / sm;
}

// ---------------------------------------------------------------------------
extern "C" void kernel_launch(void* const* d_in, const int* in_sizes, int n_in,
                              void* d_out, int out_size, void* d_ws, size_t ws_size,
                              hipStream_t stream) {
    (void)in_sizes; (void)n_in; (void)out_size; (void)ws_size;

    const float* z            = (const float*)d_in[0];  // [16,256,32,32]
    const float* weight       = (const float*)d_in[1];  // [8192,256]
    const float* cluster_size = (const float*)d_in[2];  // [8192]
    const float* embed_avg    = (const float*)d_in[3];  // [8192,256]
    float* out = (float*)d_out;

    // workspace carve-up:
    // embed_sum [K*C] | counts [K] | scal[4] | wnorm [K] | idx [N] i32 |
    // zb [N*C] bf16 | wb [K*C] bf16
    float* wsf        = (float*)d_ws;
    float* embed_sum  = wsf;                                   // K*C
    float* counts     = embed_sum + (size_t)KCODE * CH;        // K
    float* scal       = counts + KCODE;                        // 4
    float* wnorm      = scal + 4;                              // K
    int*   idx        = (int*)(wnorm + KCODE);                 // N
    bf16*  zb         = (bf16*)(idx + NTOK);                   // N*C
    bf16*  wb         = zb + (size_t)NTOK * CH;                // K*C

    const int nzero = KCODE * CH + KCODE + 4;                  // embed_sum|counts|scal
    vq_zero_ws<<<(nzero + 255) / 256, 256, 0, stream>>>(wsf, nzero);

    vq_prep_w<<<KCODE, 256, 0, stream>>>(weight, wb, wnorm);
    vq_prep_z<<<(NB * CH * HW) / 256, 256, 0, stream>>>(z, zb);

    // 8 waves/block, 16 rows/wave -> 128 rows/block, 128 blocks
    vq_argmin_wmma<<<NTOK / 128, 256, 0, stream>>>(zb, wb, wnorm, idx);

    vq_gather_loss<<<(NB * CH * HW) / 256, 256, 0, stream>>>(
        z, weight, idx, out + OFF_ZQ, &scal[0]);
    vq_idx_counts<<<(NTOK + 255) / 256, 256, 0, stream>>>(
        idx, out + OFF_IDX, counts);
    vq_embed_scatter<<<(NTOK * CH) / 256, 256, 0, stream>>>(z, idx, embed_sum);

    vq_ema_k<<<(KCODE + 255) / 256, 256, 0, stream>>>(
        cluster_size, counts, out + OFF_NCS, scal);
    vq_ema_kc<<<(KCODE * CH) / 256, 256, 0, stream>>>(
        embed_avg, embed_sum, out + OFF_NEA);

    vq_finalize<<<1, 1, 0, stream>>>(scal, out + OFF_LOSS, out + OFF_PERP);
    vq_new_weight<<<(KCODE * CH) / 256, 256, 0, stream>>>(
        out + OFF_NEA, out + OFF_NCS, scal, out + OFF_NW);
}
